// GAT_2284922602200
// MI455X (gfx1250) — matrix operand order repaired and checked by reading, not compiled, not run on hardware
//
#include <hip/hip_runtime.h>
#include <cstdint>

typedef _Float16 v16h __attribute__((ext_vector_type(16)));
typedef _Float16 v8h  __attribute__((ext_vector_type(8)));
typedef float    v8f  __attribute__((ext_vector_type(8)));

#define NN    40000
#define EE    640000
#define HIDD  128
#define NHEAD 4
#define CH    32
#define NGRAPH 16

// ---- weight conversion f32 -> f16 in WMMA B-fragment ("swizzled") order -----
// dst[((k/32)*128 + col)*32 + ((k>>4)&1)*16 + (k&15)] = src[k*128+col]
// so a lane's 16-element B fragment for (ktile, col, lanegroup) is contiguous 32B.
__global__ void wconv_swz_kernel(const float* __restrict__ src, _Float16* __restrict__ dst,
                                 int Krows, int Kpad) {
    int i = blockIdx.x * blockDim.x + threadIdx.x;      // over Kpad*128
    if (i >= Kpad * HIDD) return;
    int k = i >> 7, col = i & 127;
    int tile = k >> 5, g = (k >> 4) & 1, j = k & 15;
    float v = (k < Krows) ? src[k * HIDD + col] : 0.0f;
    dst[((tile * HIDD + col) << 5) + g * 16 + j] = (_Float16)v;
}

// ---------------- generic fill ----------------------------------------------
__global__ void fill_u32_kernel(unsigned* __restrict__ p, unsigned val, int n) {
    int i = blockIdx.x * blockDim.x + threadIdx.x;
    if (i < n) p[i] = val;
}

// init rows of [rows,128] to bias vector (h_gat starts at gat_b)
__global__ void init_bias_rows_kernel(float* __restrict__ dst, const float* __restrict__ bias, int rows) {
    int i = blockIdx.x * blockDim.x + threadIdx.x;
    if (i >= rows * HIDD) return;
    dst[i] = bias[i & (HIDD - 1)];
}

// ---------------- WMMA GEMM: out[M,128] = act(Aeff[M,K] @ W + bias) ----------
// Block = 256 threads = 8 waves; one 16-row tile, each wave owns a 16-col tile.
// A tile staged to LDS as f16 (CONCAT splices pos[*,3] at k=128..130, zero pad).
// Wswz is the fragment-ordered f16 weight buffer produced by wconv_swz_kernel.
// Grid must be exactly M/16 blocks (M multiple of 16) so EXEC is all-ones.
template <int KPAD, bool CONCAT>
__global__ __launch_bounds__(256)
void gemm_wmma_kernel(const float* __restrict__ A, const float* __restrict__ pos,
                      const _Float16* __restrict__ Wswz, const float* __restrict__ bias,
                      float* __restrict__ out, int relu) {
    __shared__ __align__(16) _Float16 As[16 * KPAD];
    const int m0 = blockIdx.x * 16;

    // ---- stage A tile (f32 -> f16) into LDS, branch cost paid once here ----
    for (int i = threadIdx.x; i < 16 * KPAD; i += 256) {
        int r = i / KPAD, k = i - r * KPAD;
        float v;
        if (CONCAT) {
            v = (k < 128) ? A[(long)(m0 + r) * HIDD + k]
                          : ((k < 131) ? pos[(m0 + r) * 3 + (k - 128)] : 0.0f);
        } else {
            v = A[(long)(m0 + r) * HIDD + k];           // K == KPAD == 128
        }
        As[r * KPAD + k] = (_Float16)v;
    }
    __syncthreads();

    const int wave = threadIdx.x >> 5;
    const int lane = threadIdx.x & 31;
    const int col0 = wave * 16;
    const int n    = lane & 15;
    const int hi   = lane >> 4;
    const _Float16* Arow = &As[(lane & 15) * KPAD];

    v8f c = {};
#pragma unroll
    for (int kb = 0; kb < KPAD; kb += 32) {
        // A fragment: element j -> k = kb + hi*16 + j
        v8h alo = *(const v8h*)&Arow[kb + hi * 16];       // k = kb+hi*16 .. +7
        v8h ahi = *(const v8h*)&Arow[kb + hi * 16 + 8];   // k = kb+hi*16+8 .. +7
        v16h a = __builtin_shufflevector(alo, ahi, 0, 1, 2, 3, 4, 5, 6, 7,
                                                   8, 9, 10, 11, 12, 13, 14, 15);
        // B fragment: one contiguous 32B chunk in swizzled weight buffer
        v16h b = *(const v16h*)&Wswz[(((kb >> 5) * HIDD + col0 + n) << 5) + hi * 16];
        c = __builtin_amdgcn_wmma_f32_16x16x32_f16(false, a, false, b, (short)0, c, false, false);
    }

#pragma unroll
    for (int r = 0; r < 8; ++r) {
        int row = m0 + r + hi * 8;                        // C/D layout
        float val = c[r] + (bias ? bias[col0 + n] : 0.0f);
        if (relu) val = fmaxf(val, 0.0f);
        out[(long)row * HIDD + col0 + n] = val;
    }
}

// ---------------- attention scores: a_src/a_dst [N,4] -------------------------
__global__ void att_score_kernel(const float* __restrict__ xw,
                                 const float* __restrict__ att_src, const float* __restrict__ att_dst,
                                 float* __restrict__ a_src, float* __restrict__ a_dst) {
    int i = blockIdx.x * blockDim.x + threadIdx.x;
    if (i >= NN * NHEAD) return;
    int node = i >> 2, h = i & 3;
    const float* xp = xw + (long)node * HIDD + h * CH;
    float s1 = 0.f, s2 = 0.f;
#pragma unroll 8
    for (int cdx = 0; cdx < CH; ++cdx) {
        float v = xp[cdx];
        s1 += v * att_src[h * CH + cdx];
        s2 += v * att_dst[h * CH + cdx];
    }
    a_src[i] = s1; a_dst[i] = s2;
}

// ordered-uint encoding so unsigned atomicMax == float max
__device__ __forceinline__ unsigned fenc(float f) {
    unsigned u = __float_as_uint(f);
    return (u & 0x80000000u) ? ~u : (u | 0x80000000u);
}
__device__ __forceinline__ float fdec(unsigned u) {
    return (u & 0x80000000u) ? __uint_as_float(u & 0x7fffffffu) : __uint_as_float(~u);
}

__device__ __forceinline__ void edge_sd(int e, const long long* ei, int& s, int& d) {
    if (e < EE) { s = (int)ei[e]; d = (int)ei[EE + e]; }
    else        { s = d = e - EE; }
}

__global__ void edge_max_kernel(const long long* __restrict__ ei,
                                const float* __restrict__ a_src, const float* __restrict__ a_dst,
                                unsigned* __restrict__ segmax) {
    int i = blockIdx.x * blockDim.x + threadIdx.x;
    if (i >= (EE + NN) * NHEAD) return;
    int e = i >> 2, h = i & 3, s, d;
    edge_sd(e, ei, s, d);
    float v = a_src[s * NHEAD + h] + a_dst[d * NHEAD + h];
    v = (v > 0.f) ? v : 0.2f * v;                 // leaky_relu
    atomicMax(&segmax[d * NHEAD + h], fenc(v));
}

__global__ void edge_expsum_kernel(const long long* __restrict__ ei,
                                   const float* __restrict__ a_src, const float* __restrict__ a_dst,
                                   const unsigned* __restrict__ segmax,
                                   float* __restrict__ alpha_e, float* __restrict__ denom) {
    int i = blockIdx.x * blockDim.x + threadIdx.x;
    if (i >= (EE + NN) * NHEAD) return;
    int e = i >> 2, h = i & 3, s, d;
    edge_sd(e, ei, s, d);
    float v = a_src[s * NHEAD + h] + a_dst[d * NHEAD + h];
    v = (v > 0.f) ? v : 0.2f * v;
    float ex = expf(v - fdec(segmax[d * NHEAD + h]));
    alpha_e[i] = ex;
    atomicAdd(&denom[d * NHEAD + h], ex);
}

__global__ void edge_aggregate_kernel(const long long* __restrict__ ei,
                                      const float* __restrict__ alpha_e, const float* __restrict__ denom,
                                      const float* __restrict__ xw, float* __restrict__ h_gat) {
    long i = (long)blockIdx.x * blockDim.x + threadIdx.x;
    if (i >= (long)(EE + NN) * HIDD) return;
    int e = (int)(i >> 7), cdx = (int)(i & 127), h = cdx >> 5, s, d;
    edge_sd(e, ei, s, d);
    float w = alpha_e[e * NHEAD + h] / (denom[d * NHEAD + h] + 1e-16f);
    atomicAdd(&h_gat[(long)d * HIDD + cdx], w * xw[(long)s * HIDD + cdx]);
}

// ---------------- residual + LayerNorm + ELU (one node per 128-thread block) --
__global__ __launch_bounds__(128)
void residual_ln_elu_kernel(float* __restrict__ h, const float* __restrict__ h_gat,
                            const float* __restrict__ h_sp, const float* __restrict__ alpha_p,
                            const float* __restrict__ ln_g, const float* __restrict__ ln_b) {
    __shared__ float sh[HIDD];
    int node = blockIdx.x, c = threadIdx.x;
    float a = alpha_p[0];
    long idx = (long)node * HIDD + c;
    float v = h[idx] + a * h_gat[idx] + (1.0f - a) * h_sp[idx];
    sh[c] = v;
    __syncthreads();
    float mu = 0.f;
    for (int j = 0; j < HIDD; ++j) mu += sh[j];
    mu *= (1.0f / HIDD);
    float var = 0.f;
    for (int j = 0; j < HIDD; ++j) { float dv = sh[j] - mu; var += dv * dv; }
    var *= (1.0f / HIDD);
    float y = ln_g[c] * (v - mu) * rsqrtf(var + 1e-5f) + ln_b[c];
    h[idx] = (y > 0.f) ? y : (expf(y) - 1.0f);    // ELU
}

// ---------------- global mean pool -------------------------------------------
__global__ void pool_kernel(const float* __restrict__ h, const long long* __restrict__ batch,
                            float* __restrict__ pool, float* __restrict__ cnt) {
    long i = (long)blockIdx.x * blockDim.x + threadIdx.x;
    if (i >= (long)NN * HIDD) return;
    int node = (int)(i >> 7), c = (int)(i & 127);
    int g = (int)batch[node];
    atomicAdd(&pool[g * HIDD + c], h[i]);
    if (c == 0) atomicAdd(&cnt[g], 1.0f);
}

__global__ void pool_div_kernel(float* __restrict__ pool, const float* __restrict__ cnt) {
    int i = blockIdx.x * blockDim.x + threadIdx.x;
    if (i >= NGRAPH * HIDD) return;
    pool[i] /= fmaxf(cnt[i >> 7], 1.0f);
}

// =============================================================================
extern "C" void kernel_launch(void* const* d_in, const int* in_sizes, int n_in,
                              void* d_out, int out_size, void* d_ws, size_t ws_size,
                              hipStream_t stream) {
    (void)in_sizes; (void)n_in; (void)out_size; (void)ws_size;
    const int ETOT = EE + NN;

    // ---- inputs (setup_inputs() insertion order, params recursively flattened)
    const float*     x     = (const float*)d_in[0];
    const float*     pos   = (const float*)d_in[1];
    const long long* ei    = (const long long*)d_in[2];
    const long long* batch = (const long long*)d_in[3];
    const float*     in_w  = (const float*)d_in[4];
    const float*     in_b  = (const float*)d_in[5];
    // layers: base 6 + 9*l: gat_w, att_src, att_dst, gat_b, sp_w, sp_b, ln_g, ln_b, alpha
    const float* out_w1 = (const float*)d_in[33];
    const float* out_b1 = (const float*)d_in[34];
    const float* out_w2 = (const float*)d_in[35];
    const float* out_b2 = (const float*)d_in[36];

    // ---- workspace carve-out
    char* p = (char*)d_ws;
    auto alloc = [&](size_t bytes) { void* r = (void*)p; p += (bytes + 255) & ~size_t(255); return r; };
    float*     h       = (float*)alloc((size_t)NN * HIDD * 4);
    float*     xw      = (float*)alloc((size_t)NN * HIDD * 4);
    float*     h_gat   = (float*)alloc((size_t)NN * HIDD * 4);
    float*     h_sp    = (float*)alloc((size_t)NN * HIDD * 4);
    float*     a_src   = (float*)alloc((size_t)NN * NHEAD * 4);
    float*     a_dst   = (float*)alloc((size_t)NN * NHEAD * 4);
    unsigned*  segmax  = (unsigned*)alloc((size_t)NN * NHEAD * 4);
    float*     denom   = (float*)alloc((size_t)NN * NHEAD * 4);
    float*     alpha_e = (float*)alloc((size_t)ETOT * NHEAD * 4);
    float*     pool    = (float*)alloc((size_t)NGRAPH * HIDD * 4);
    float*     cnt     = (float*)alloc((size_t)NGRAPH * 4);
    float*     tmp1    = (float*)alloc((size_t)NGRAPH * HIDD * 4);
    _Float16*  w_in    = (_Float16*)alloc((size_t)128 * HIDD * 2);
    _Float16*  w_gat[3]; _Float16* w_sp[3];
    for (int l = 0; l < 3; ++l) {
        w_gat[l] = (_Float16*)alloc((size_t)128 * HIDD * 2);
        w_sp[l]  = (_Float16*)alloc((size_t)160 * HIDD * 2);   // K=131 padded to 160
    }
    _Float16* w_o1 = (_Float16*)alloc((size_t)128 * HIDD * 2);
    _Float16* w_o2 = (_Float16*)alloc((size_t)128 * HIDD * 2);

    const int T = 256;
    auto blk = [&](long n) { return (unsigned)((n + T - 1) / T); };

    // ---- weight conversion to swizzled f16 (tiny; per-call for determinism)
    wconv_swz_kernel<<<blk(128 * HIDD), T, 0, stream>>>(in_w, w_in, 128, 128);
    for (int l = 0; l < 3; ++l) {
        const float* gat_w = (const float*)d_in[6 + 9 * l + 0];
        const float* sp_w  = (const float*)d_in[6 + 9 * l + 4];
        wconv_swz_kernel<<<blk(128 * HIDD), T, 0, stream>>>(gat_w, w_gat[l], 128, 128);
        wconv_swz_kernel<<<blk(160 * HIDD), T, 0, stream>>>(sp_w, w_sp[l], 131, 160);
    }
    wconv_swz_kernel<<<blk(128 * HIDD), T, 0, stream>>>(out_w1, w_o1, 128, 128);
    wconv_swz_kernel<<<blk(128 * HIDD), T, 0, stream>>>(out_w2, w_o2, 128, 128);

    // ---- h = x @ in_w + in_b
    gemm_wmma_kernel<128, false><<<NN / 16, 256, 0, stream>>>(x, nullptr, w_in, in_b, h, 0);

    // ---- layers
    for (int l = 0; l < 3; ++l) {
        const float* att_s  = (const float*)d_in[6 + 9 * l + 1];
        const float* att_d  = (const float*)d_in[6 + 9 * l + 2];
        const float* gat_b  = (const float*)d_in[6 + 9 * l + 3];
        const float* sp_b   = (const float*)d_in[6 + 9 * l + 5];
        const float* ln_g   = (const float*)d_in[6 + 9 * l + 6];
        const float* ln_b   = (const float*)d_in[6 + 9 * l + 7];
        const float* alphap = (const float*)d_in[6 + 9 * l + 8];

        // xw = h @ gat_w (no bias)
        gemm_wmma_kernel<128, false><<<NN / 16, 256, 0, stream>>>(h, nullptr, w_gat[l], nullptr, xw, 0);

        att_score_kernel<<<blk((long)NN * NHEAD), T, 0, stream>>>(xw, att_s, att_d, a_src, a_dst);

        // segment softmax + aggregation
        fill_u32_kernel<<<blk((long)NN * NHEAD), T, 0, stream>>>(segmax, 0x007FFFFFu, NN * NHEAD); // enc(-inf)
        fill_u32_kernel<<<blk((long)NN * NHEAD), T, 0, stream>>>((unsigned*)denom, 0u, NN * NHEAD);
        init_bias_rows_kernel<<<blk((long)NN * HIDD), T, 0, stream>>>(h_gat, gat_b, NN);

        edge_max_kernel<<<blk((long)ETOT * NHEAD), T, 0, stream>>>(ei, a_src, a_dst, segmax);
        edge_expsum_kernel<<<blk((long)ETOT * NHEAD), T, 0, stream>>>(ei, a_src, a_dst, segmax, alpha_e, denom);
        edge_aggregate_kernel<<<blk((long)ETOT * HIDD), T, 0, stream>>>(ei, alpha_e, denom, xw, h_gat);

        // h_sp = concat(h_gat, pos) @ sp_w + sp_b   (K=131 -> KPAD=160)
        gemm_wmma_kernel<160, true><<<NN / 16, 256, 0, stream>>>(h_gat, pos, w_sp[l], sp_b, h_sp, 0);

        // h = elu(LN(h + a*h_gat + (1-a)*h_sp))
        residual_ln_elu_kernel<<<NN, 128, 0, stream>>>(h, h_gat, h_sp, alphap, ln_g, ln_b);
    }

    // ---- global mean pool
    fill_u32_kernel<<<blk(NGRAPH * HIDD), T, 0, stream>>>((unsigned*)pool, 0u, NGRAPH * HIDD);
    fill_u32_kernel<<<1, T, 0, stream>>>((unsigned*)cnt, 0u, NGRAPH);
    pool_kernel<<<blk((long)NN * HIDD), T, 0, stream>>>(h, batch, pool, cnt);
    pool_div_kernel<<<blk(NGRAPH * HIDD), T, 0, stream>>>(pool, cnt);

    // ---- output MLP (single 16-row WMMA tile each)
    gemm_wmma_kernel<128, false><<<1, 256, 0, stream>>>(pool, nullptr, w_o1, out_b1, tmp1, 1);
    gemm_wmma_kernel<128, false><<<1, 256, 0, stream>>>(tmp1, nullptr, w_o2, out_b2, (float*)d_out, 0);
}
